// FKANLinear_34102040330855
// MI455X (gfx1250) — matrix operand order repaired and checked by reading, not compiled, not run on hardware
//
#include <hip/hip_runtime.h>
#include <hip/hip_bf16.h>
#include <math.h>

typedef _Float16 v8h  __attribute__((ext_vector_type(8)));
typedef _Float16 v16h __attribute__((ext_vector_type(16)));
typedef float    v8f  __attribute__((ext_vector_type(8)));

#define BB    32768
#define INF_  64
#define OUTF  32
#define KPB   46      // 1 (x) + 11 bspline + 4 taylor + 5 jacobi + 5 cheby + 16 fourier + 4 wavelet
#define KPAIR 96      // two inputs (92) padded to 3x32
#define KTOT  3072    // 32 pairs * 96
#define BM    128
#define AS    104     // A panel stride (halfs), 208B rows: 16B aligned, good bank spread
#define BS    104     // B panel stride (halfs)

// workspace byte offsets
#define WS_XMIN   0        // 64 f32
#define WS_XMAX   256      // 64 f32
#define WS_GRID   512      // 64*16 f32 (15 knots + 1 pad)
#define WS_WAVA   4608     // 64*4 f32  softplus(scale)+1e-6
#define WS_BIAS   5632     // 32 f32    alpha*bias
#define WS_WALL   8192     // 32*3072 f16 fused weights [o][k]

__device__ __forceinline__ float softplusf(float z) {
    return z > 20.f ? z : log1pf(expf(z));
}

// ---------------- K0: column min/max of x ----------------
__global__ __launch_bounds__(256) void k_colstat(const float* __restrict__ x, void* ws) {
    __shared__ float smn[256], smx[256];
    int col = blockIdx.x;
    float mn = 3.4e38f, mx = -3.4e38f;
    for (int r = threadIdx.x; r < BB; r += 256) {
        float v = x[(size_t)r * INF_ + col];
        mn = fminf(mn, v); mx = fmaxf(mx, v);
    }
    smn[threadIdx.x] = mn; smx[threadIdx.x] = mx;
    __syncthreads();
    for (int s = 128; s > 0; s >>= 1) {
        if (threadIdx.x < s) {
            smn[threadIdx.x] = fminf(smn[threadIdx.x], smn[threadIdx.x + s]);
            smx[threadIdx.x] = fmaxf(smx[threadIdx.x], smx[threadIdx.x + s]);
        }
        __syncthreads();
    }
    if (threadIdx.x == 0) {
        ((float*)((char*)ws + WS_XMIN))[col] = smn[0];
        ((float*)((char*)ws + WS_XMAX))[col] = smx[0];
    }
}

// ---------------- K1: knots, wavelet scales, bias ----------------
__global__ __launch_bounds__(256) void k_prep_misc(const float* __restrict__ scale_logit,
                                                   const float* __restrict__ base_bias,
                                                   const float* __restrict__ alpha_logit,
                                                   void* ws) {
    int t = threadIdx.x;
    const float* xmin = (const float*)((char*)ws + WS_XMIN);
    const float* xmax = (const float*)((char*)ws + WS_XMAX);
    float* grid  = (float*)((char*)ws + WS_GRID);
    float* wava  = (float*)((char*)ws + WS_WAVA);
    float* biasE = (float*)((char*)ws + WS_BIAS);

    if (t < 64) {
        float mn = xmin[t], mx = xmax[t];
        if (mx - mn < 1e-8f) { mn -= 0.5f; mx += 0.5f; }
        float step = (mx - mn) * 0.125f;   // G = 8
        float* g = grid + t * 16;
        g[0] = mn; g[1] = mn; g[2] = mn;               // P=3 repeats
        #pragma unroll
        for (int j = 0; j <= 8; ++j) g[3 + j] = mn + step * (float)j;
        g[12] = mx; g[13] = mx; g[14] = mx; g[15] = mx; // tail repeats + pad
    }
    wava[t] = softplusf(scale_logit[t]) + 1e-6f;        // t < 256 == 64*4
    if (t < 32) biasE[t] = softplusf(alpha_logit[0]) * base_bias[t];
}

// ---------------- K2: fused f16 weight matrix Wall[o][3072] ----------------
__global__ __launch_bounds__(256) void k_prep_w(const float* __restrict__ base_v,
                                                const float* __restrict__ base_g,
                                                const float* __restrict__ bs_c,
                                                const float* __restrict__ ty_c,
                                                const float* __restrict__ jc_c,
                                                const float* __restrict__ ch_c,
                                                const float* __restrict__ fo_c,
                                                const float* __restrict__ wv_c,
                                                const float* __restrict__ gains,
                                                const float* __restrict__ alpha_logit,
                                                const float* __restrict__ beta_logit,
                                                const float* __restrict__ mix_logits,
                                                void* ws) {
    int gt = blockIdx.x * 256 + threadIdx.x;   // 0..2047
    int o = gt >> 6, i = gt & 63;
    int oi = o * 64 + i;

    float vn = 0.f;
    for (int ii = 0; ii < 64; ++ii) { float v = base_v[o * 64 + ii]; vn += v * v; }
    vn = sqrtf(vn);
    float W = base_g[o] * base_v[oi] / vn;
    float alpha = softplusf(alpha_logit[0]);
    float beta  = softplusf(beta_logit[0]);

    // softmax(mix/TEMP), TEMP=2
    float m[6], mxl = -3.4e38f;
    #pragma unroll
    for (int f = 0; f < 6; ++f) { m[f] = mix_logits[oi * 6 + f] * 0.5f; mxl = fmaxf(mxl, m[f]); }
    float se = 0.f;
    #pragma unroll
    for (int f = 0; f < 6; ++f) { m[f] = expf(m[f] - mxl); se += m[f]; }
    float sc[6];
    #pragma unroll
    for (int f = 0; f < 6; ++f) sc[f] = beta * softplusf(gains[f]) * (m[f] / se);

    _Float16* wall = (_Float16*)((char*)ws + WS_WALL) + (size_t)o * KTOT;
    int p  = i >> 1;
    int kb = p * KPAIR + (i & 1) * KPB;

    wall[kb + 0] = (_Float16)(alpha * W);
    const float* c;
    c = bs_c + (size_t)oi * 11;
    #pragma unroll
    for (int j = 0; j < 11; ++j) wall[kb + 1 + j]  = (_Float16)(sc[0] * c[j]);
    c = ty_c + (size_t)oi * 4;
    #pragma unroll
    for (int j = 0; j < 4;  ++j) wall[kb + 12 + j] = (_Float16)(sc[1] * c[j]);
    c = jc_c + (size_t)oi * 5;
    #pragma unroll
    for (int j = 0; j < 5;  ++j) wall[kb + 16 + j] = (_Float16)(sc[2] * c[j]);
    c = ch_c + (size_t)oi * 5;
    #pragma unroll
    for (int j = 0; j < 5;  ++j) wall[kb + 21 + j] = (_Float16)(sc[3] * c[j]);
    c = fo_c + (size_t)oi * 16;
    #pragma unroll
    for (int j = 0; j < 16; ++j) wall[kb + 26 + j] = (_Float16)(sc[4] * c[j]);
    c = wv_c + (size_t)oi * 4;
    #pragma unroll
    for (int j = 0; j < 4;  ++j) wall[kb + 42 + j] = (_Float16)(sc[5] * c[j]);

    if (!(i & 1)) {
        #pragma unroll
        for (int j = 0; j < 4; ++j) wall[p * KPAIR + 92 + j] = (_Float16)0.f;
    }
}

// ---------------- K3: fused basis-gen + WMMA GEMM ----------------
__global__ __launch_bounds__(256) void k_fkan(const float* __restrict__ x,
                                              const float* __restrict__ shift,
                                              const void* __restrict__ ws,
                                              float* __restrict__ out) {
    __shared__ __align__(32) _Float16 ldsA[BM * AS];
    __shared__ __align__(32) _Float16 ldsB[32 * BS];
    __shared__ float ldsG[64 * 16];
    __shared__ float ldsWa[256];
    __shared__ float ldsSh[256];
    __shared__ float ldsBias[32];

    const int tid  = threadIdx.x;
    const int lane = tid & 31;
    const int wave = tid >> 5;
    const int r0   = blockIdx.x * BM;

    const float*    wsGrid = (const float*)((const char*)ws + WS_GRID);
    const float*    wsWa   = (const float*)((const char*)ws + WS_WAVA);
    const float*    wsBias = (const float*)((const char*)ws + WS_BIAS);
    const _Float16* wall   = (const _Float16*)((const char*)ws + WS_WALL);

    for (int u = tid; u < 64 * 16; u += 256) ldsG[u] = wsGrid[u];
    ldsWa[tid] = wsWa[tid];
    ldsSh[tid] = shift[tid];
    if (tid < 32) ldsBias[tid] = wsBias[tid];
    __syncthreads();

    v8f acc0 = {0.f,0.f,0.f,0.f,0.f,0.f,0.f,0.f};
    v8f acc1 = {0.f,0.f,0.f,0.f,0.f,0.f,0.f,0.f};

    const int row   = tid >> 1;
    const int which = tid & 1;
    const int wrow0 = wave * 16;
    const int al    = lane & 15;
    const int ah    = lane >> 4;   // 0 / 1
    const int hi8   = ah * 8;      // A-fragment K offset (halfs)
    const int bkh   = ah * 8;      // B-fragment K offset (halfs)

    for (int p = 0; p < 32; ++p) {
        // ---- stage B panel: Wall[n][p*96 .. p*96+96) -> ldsB[n][0..96)
        {
            const _Float16* src = wall + p * KPAIR;
            #pragma unroll
            for (int rpt = 0; rpt < 2; ++rpt) {
                int u = tid + rpt * 256;
                if (u < 32 * 12) {
                    int n = u / 12, c = u % 12;
                    *(v8h*)&ldsB[n * BS + c * 8] =
                        *(const v8h*)(src + (size_t)n * KTOT + c * 8);
                }
            }
            if (p + 1 < 32)  // warm next panel into cache (global_prefetch_b8)
                __builtin_prefetch(wall + (p + 1) * KPAIR + (size_t)(lane) * KTOT, 0, 1);
        }
        // ---- compute 46 basis values for one (row, i) point into ldsA
        {
            int i = 2 * p + which;
            float xv = x[(size_t)(r0 + row) * INF_ + i];
            _Float16* dst = &ldsA[row * AS + which * KPB];

            dst[0] = (_Float16)xv;                       // identity slot (alpha*W)

            // b-spline, Cox-de Boor degree 3, 15 clamped knots
            float g[15];
            #pragma unroll
            for (int t = 0; t < 15; ++t) g[t] = ldsG[i * 16 + t];
            float bas[14];
            #pragma unroll
            for (int t = 0; t < 14; ++t)
                bas[t] = (xv >= g[t] && xv < g[t + 1]) ? 1.f : 0.f;
            if (xv == g[14]) {
                #pragma unroll
                for (int t = 0; t < 13; ++t) bas[t] = 0.f;
                bas[13] = 1.f;
            }
            #pragma unroll
            for (int r = 1; r <= 3; ++r) {
                #pragma unroll
                for (int t = 0; t < 13; ++t) {
                    if (t < 14 - r) {
                        float ld = fmaxf(g[t + r] - g[t], 1e-12f);
                        float rd = fmaxf(g[t + r + 1] - g[t + 1], 1e-12f);
                        bas[t] = (xv - g[t]) / ld * bas[t]
                               + (g[t + r + 1] - xv) / rd * bas[t + 1];
                    }
                }
            }
            #pragma unroll
            for (int j = 0; j < 11; ++j) dst[1 + j] = (_Float16)bas[j];

            // taylor: x^k / k!
            float x2 = xv * xv;
            dst[12] = (_Float16)1.f;
            dst[13] = (_Float16)xv;
            dst[14] = (_Float16)(0.5f * x2);
            dst[15] = (_Float16)(x2 * xv * (1.f / 6.f));

            // jacobi (a=b=1), normalized recurrence
            float jm2 = 1.f, jm1 = 1.41421356237f * xv;
            dst[16] = (_Float16)jm2;
            dst[17] = (_Float16)jm1;
            #pragma unroll
            for (int n = 2; n <= 4; ++n) {
                float k  = (float)(n - 1);
                float A1 = 2.f * k + 2.f;
                float A2 = 2.f * (k + 1.f) * (k + 3.f) * (A1 + 1.f);
                float A3 = (A1 + 1.f) * ((A1 + 2.f) * A1 * xv);
                float A4 = 2.f * (k + 1.f) * (k + 1.f) * (A1 + 2.f);
                float Jn = (A3 * jm1 - A4 * jm2) / A2;
                float v  = Jn * rsqrtf((float)n + 1.f);
                dst[16 + n] = (_Float16)v;
                jm2 = jm1; jm1 = v;
            }

            // chebyshev
            float xc = __builtin_isfinite(xv) ? xv : 0.f;
            xc = fminf(fmaxf(xc, -1e6f), 1e6f);
            float t0 = 1.f, t1 = xc;
            dst[21] = (_Float16)1.f;
            dst[22] = (_Float16)(t1 * 0.70710678f);
            #pragma unroll
            for (int n = 2; n <= 4; ++n) {
                float tn = 2.f * xc * t1 - t0;
                dst[21 + n] = (_Float16)(tn * rsqrtf((float)n + 1.f));
                t0 = t1; t1 = tn;
            }

            // fourier: one sincos + angle-addition recurrence, scale 1/sqrt(2F)=0.25
            float s1, c1;
            __sincosf(xv, &s1, &c1);
            float ck = c1, sk = s1, ckm = 1.f, skm = 0.f;
            #pragma unroll
            for (int kk = 0; kk < 8; ++kk) {
                dst[26 + kk] = (_Float16)(ck * 0.25f);
                dst[34 + kk] = (_Float16)(sk * 0.25f);
                float cn = 2.f * c1 * ck - ckm;
                float sn = 2.f * c1 * sk - skm;
                ckm = ck; skm = sk; ck = cn; sk = sn;
            }

            // mexican-hat wavelet
            #pragma unroll
            for (int c = 0; c < 4; ++c) {
                float aw = ldsWa[i * 4 + c];
                float u  = (xv - ldsSh[i * 4 + c]) / aw;
                float u2 = u * u;
                dst[42 + c] = (_Float16)((u2 - 1.f) * __expf(-0.5f * u2));
            }

            if (!which) {   // zero the 4-half pad of this pair's row
                _Float16* pz = &ldsA[row * AS + 92];
                pz[0] = (_Float16)0.f; pz[1] = (_Float16)0.f;
                pz[2] = (_Float16)0.f; pz[3] = (_Float16)0.f;
            }
        }
        __syncthreads();

        // ---- 3 K-steps x 2 N-halves of v_wmma_f32_16x16x32_f16
        #pragma unroll
        for (int ks = 0; ks < 3; ++ks) {
            const int k0 = ks * 32;
            const _Float16* pa = &ldsA[(wrow0 + al) * AS + k0 + hi8];
            v8h a_lo = *(const v8h*)pa;
            v8h a_hi = *(const v8h*)(pa + 16);
            v16h a = __builtin_shufflevector(a_lo, a_hi,
                     0,1,2,3,4,5,6,7,8,9,10,11,12,13,14,15);

            const _Float16* pb0 = &ldsB[al * BS + k0 + bkh];
            v8h b0l = *(const v8h*)pb0;
            v8h b0h = *(const v8h*)(pb0 + 16);
            v16h b0 = __builtin_shufflevector(b0l, b0h,
                      0,1,2,3,4,5,6,7,8,9,10,11,12,13,14,15);

            const _Float16* pb1 = &ldsB[(16 + al) * BS + k0 + bkh];
            v8h b1l = *(const v8h*)pb1;
            v8h b1h = *(const v8h*)(pb1 + 16);
            v16h b1 = __builtin_shufflevector(b1l, b1h,
                      0,1,2,3,4,5,6,7,8,9,10,11,12,13,14,15);

            acc0 = __builtin_amdgcn_wmma_f32_16x16x32_f16(
                       false, a, false, b0, (short)0, acc0, false, false);
            acc1 = __builtin_amdgcn_wmma_f32_16x16x32_f16(
                       false, a, false, b1, (short)0, acc1, false, false);
        }
        __syncthreads();
    }

    // ---- epilogue: C layout -> global (f32), add alpha*bias
    {
        int m0 = wrow0 + ah * 8;
        float bb0 = ldsBias[al];
        float bb1 = ldsBias[16 + al];
        #pragma unroll
        for (int j = 0; j < 8; ++j) {
            int ro = r0 + m0 + j;
            out[(size_t)ro * OUTF + al]      = acc0[j] + bb0;
            out[(size_t)ro * OUTF + 16 + al] = acc1[j] + bb1;
        }
    }
}

extern "C" void kernel_launch(void* const* d_in, const int* in_sizes, int n_in,
                              void* d_out, int out_size, void* d_ws, size_t ws_size,
                              hipStream_t stream) {
    (void)in_sizes; (void)n_in; (void)out_size; (void)ws_size;
    const float* x  = (const float*)d_in[0];
    const float* bv = (const float*)d_in[1];
    const float* bg = (const float*)d_in[2];
    const float* bb = (const float*)d_in[3];
    const float* cb = (const float*)d_in[4];
    const float* ct = (const float*)d_in[5];
    const float* cj = (const float*)d_in[6];
    const float* cc = (const float*)d_in[7];
    const float* cf = (const float*)d_in[8];
    const float* cw = (const float*)d_in[9];
    const float* wsl = (const float*)d_in[10];
    const float* wsh = (const float*)d_in[11];
    const float* gn  = (const float*)d_in[12];
    const float* al  = (const float*)d_in[13];
    const float* be  = (const float*)d_in[14];
    const float* ml  = (const float*)d_in[15];

    k_colstat<<<64, 256, 0, stream>>>(x, d_ws);
    k_prep_misc<<<1, 256, 0, stream>>>(wsl, bb, al, d_ws);
    k_prep_w<<<8, 256, 0, stream>>>(bv, bg, cb, ct, cj, cc, cf, cw, gn, al, be, ml, d_ws);
    k_fkan<<<BB / BM, 256, 0, stream>>>(x, wsh, d_ws, (float*)d_out);
}
